// DynamicConv_53910429499723
// MI455X (gfx1250) — matrix operand order repaired and checked by reading, not compiled, not run on hardware
//
#include <hip/hip_runtime.h>
#include <hip/hip_bf16.h>

typedef __attribute__((ext_vector_type(16))) __bf16 v16bf;
typedef __attribute__((ext_vector_type(8)))  __bf16 v8bf;
typedef __attribute__((ext_vector_type(8)))  float  v8f;

#define BATCH   32
#define CCH     256
#define HWSZ    4096        // 64*64
#define EE      16
#define NCHUNK  8           // 256 ic / 32 per chunk
#define QTOT    72          // K32-chunks total = NCHUNK * 9 taps
#define ICPAD   40          // LDS ic stride (bf16): 80B -> 16B aligned, 20-dword bank stride (conflict-free over 16 cols)
#define AFRAG_ELEMS (16*QTOT*32*16)   // 589824 bf16 per hi/lo array

#ifndef __has_builtin
#define __has_builtin(x) 0
#endif
#if __has_builtin(__builtin_amdgcn_global_load_async_to_lds_b128)
#define HAVE_ASYNC 1
#else
#define HAVE_ASYNC 0
#endif

// The builtin's param 1 is 'int __attribute__((vector_size(16))) __device__*'
// (addrspace(1) int4*); param 2 is the LDS-side (addrspace(3)) pointer.
typedef int v4i_gcc __attribute__((vector_size(16)));
typedef v4i_gcc __attribute__((address_space(1))) GV4;
typedef v4i_gcc __attribute__((address_space(3))) LV4;

__device__ __forceinline__ void wait_async_all() {
#if __has_builtin(__builtin_amdgcn_s_wait_asynccnt)
    __builtin_amdgcn_s_wait_asynccnt(0);
#elif HAVE_ASYNC
    asm volatile("s_wait_asynccnt 0x0" ::: "memory");
#endif
}

// ---------------------------------------------------------------------------
// Kernel 1: weight = sum_e conv_para[e], split into bf16 hi/lo, stored directly
// in WMMA A-fragment order: flat idx = ((g*QTOT + q)*32 + lane)*16 + j
// Per ISA 7.12.2 (16-bit A 16x32): lane<16 holds M=lane, lane>=16 holds M=lane-16
// and this code uses K=j(+16*half) (contiguous order, matching B).
// ---------------------------------------------------------------------------
__global__ void prep_weight(const float* __restrict__ cp,
                            __bf16* __restrict__ awh, __bf16* __restrict__ awl) {
    int idx = blockIdx.x * 256 + threadIdx.x;
    if (idx >= AFRAG_ELEMS) return;
    int j    = idx & 15;
    int lane = (idx >> 4) & 31;
    int q    = (idx >> 9) % QTOT;
    int g    = idx / (QTOT * 512);
    int m    = lane & 15;
    int half = lane >> 4;
    int k = j + 16 * half;   // K index within 0..31
    int icc = q / 9, f = q % 9, kh = f / 3, kw = f % 3;
    int ic = icc * 32 + k;
    int oc = g * 16 + m;
    float s = 0.f;
    #pragma unroll
    for (int e = 0; e < EE; ++e)
        s += cp[(((e * CCH + oc) * CCH + ic) * 3 + kh) * 3 + kw];
    __bf16 h = (__bf16)s;
    __bf16 l = (__bf16)(s - (float)h);
    awh[idx] = h;
    awl[idx] = l;
}

// Load 16 contiguous bf16 (32B) from LDS as two 16B-aligned halves.
__device__ __forceinline__ v16bf ld_lds_v16bf(const __bf16* p) {
    v8bf a = *(const v8bf*)p;
    v8bf b = *(const v8bf*)(p + 8);
    return __builtin_shufflevector(a, b, 0,1,2,3,4,5,6,7,8,9,10,11,12,13,14,15);
}

// Stage one ic-chunk's interior (in-image rows, cols 0..63) into an LDS f32
// buffer laid out [ic][r][64], as 16B lane-copies. Out-of-image rows are
// skipped (zeroing happens in the conversion pass).
__device__ __forceinline__ void stage_chunk(const float* __restrict__ x, float* fbuf,
                                            int b, int icc, int y0, int tid) {
    #pragma unroll
    for (int i = 0; i < 8; ++i) {
        int u   = tid + 256 * i;       // 2048 x 16B chunks
        int c16 = u & 15;              // 16B chunk within the 64-col row
        int r   = (u >> 4) & 3;
        int ic  = u >> 6;
        int yr  = y0 - 1 + r;
        if ((unsigned)yr < 64u) {
            const float* g = x + (((size_t)b * CCH + icc * 32 + ic) << 12) + (yr << 6) + (c16 << 2);
            float*       l = fbuf + ((ic * 4 + r) << 6) + (c16 << 2);
#if HAVE_ASYNC
            __builtin_amdgcn_global_load_async_to_lds_b128((GV4*)g, (LV4*)l, 0, 0);
#else
            *(float4*)l = *(const float4*)g;
#endif
        }
    }
}

// ---------------------------------------------------------------------------
// Kernel 2: implicit-GEMM 3x3 conv via split-bf16 WMMA, double-buffered
// async staging. Block: 256 threads (8 waves); tile 64 oc x 128 px (2 rows).
// ---------------------------------------------------------------------------
__global__ __launch_bounds__(256) void conv_wmma(
        const float* __restrict__ x,
        const __bf16* __restrict__ awh, const __bf16* __restrict__ awl,
        float* __restrict__ y) {
    __shared__ float  fta[32 * 4 * 64];          // 32 KB double buffer A
    __shared__ float  ftb[32 * 4 * 64];          // 32 KB double buffer B
    __shared__ __bf16 shi[4 * 66 * ICPAD];       // 21 KB hi tile
    __shared__ __bf16 slo[4 * 66 * ICPAD];       // 21 KB lo tile

    const int tid  = threadIdx.x;
    const int lane = tid & 31;
    const int wave = tid >> 5;
    const int wm   = wave & 1;
    const int wp   = wave >> 1;
    const int b    = blockIdx.x >> 5;
    const int y0   = (blockIdx.x & 31) * 2;        // first of 2 output rows
    const int ocb  = blockIdx.y * 64 + wm * 32;    // wave's oc base
    const int gb   = blockIdx.y * 4 + wm * 2;      // wave's A-fragment group base

    const int n     = lane & 15;          // B column (pixel within 16)
    const int khalf = (lane >> 4) << 4;   // K half: 16*(lane/16), contiguous order as in A

    v8f acc00 = {}, acc01 = {}, acc10 = {}, acc11 = {};

    stage_chunk(x, fta, b, 0, y0, tid);   // prologue: chunk 0 in flight

    for (int icc = 0; icc < NCHUNK; ++icc) {
        float* cur = (icc & 1) ? ftb : fta;
        float* nxt = (icc & 1) ? fta : ftb;

        wait_async_all();      // this wave's async copies into `cur` done
        __syncthreads();       // all waves' copies visible; prev taps done

        // ---- convert f32 tile -> bf16 hi/lo tiles, zeroing the halo ----
        for (int t = tid; t < 32 * 4 * 66; t += 256) {
            int c  = t % 66;
            int r  = (t / 66) & 3;
            int ic = t / 264;
            int yr = y0 - 1 + r;
            int xc = c - 1;
            float v = 0.f;
            if ((unsigned)yr < 64u && (unsigned)xc < 64u)
                v = cur[((ic * 4 + r) << 6) + xc];
            __bf16 h = (__bf16)v;
            __bf16 l = (__bf16)(v - (float)h);
            int o = (r * 66 + c) * ICPAD + ic;
            shi[o] = h;
            slo[o] = l;
        }

        if (icc + 1 < NCHUNK)                     // overlap next fetch with WMMAs
            stage_chunk(x, nxt, b, icc + 1, y0, tid);

        __syncthreads();       // bf16 tiles ready for all waves

        // ---- 9 taps of K=32 each ----
        #pragma unroll
        for (int f = 0; f < 9; ++f) {
            const int kh = f / 3, kw = f % 3;
            const int q  = icc * 9 + f;

            // A fragments (weights, pre-permuted) from global/L2
            const size_t a0 = (size_t)(((gb + 0) * QTOT + q) * 32 + lane) << 4;
            const size_t a1 = (size_t)(((gb + 1) * QTOT + q) * 32 + lane) << 4;
            v16bf A0h = *(const v16bf*)(awh + a0);
            v16bf A1h = *(const v16bf*)(awh + a1);
            v16bf A0l = *(const v16bf*)(awl + a0);
            v16bf A1l = *(const v16bf*)(awl + a1);

            // B fragments (input patches) from LDS: 16 contiguous ic per lane
            int px0  = wp * 32 + n;
            int off0 = (((px0 >> 6) + kh) * 66 + (px0 & 63) + kw) * ICPAD + khalf;
            v16bf B0h = ld_lds_v16bf(shi + off0);
            v16bf B0l = ld_lds_v16bf(slo + off0);
            int px1  = px0 + 16;
            int off1 = (((px1 >> 6) + kh) * 66 + (px1 & 63) + kw) * ICPAD + khalf;
            v16bf B1h = ld_lds_v16bf(shi + off1);
            v16bf B1l = ld_lds_v16bf(slo + off1);

            // split-bf16 fp32 emulation: hi*hi + hi*lo + lo*hi (fp32 accum)
            acc00 = __builtin_amdgcn_wmma_f32_16x16x32_bf16(false, A0h, false, B0h, (short)0, acc00, false, false);
            acc00 = __builtin_amdgcn_wmma_f32_16x16x32_bf16(false, A0h, false, B0l, (short)0, acc00, false, false);
            acc00 = __builtin_amdgcn_wmma_f32_16x16x32_bf16(false, A0l, false, B0h, (short)0, acc00, false, false);

            acc01 = __builtin_amdgcn_wmma_f32_16x16x32_bf16(false, A0h, false, B1h, (short)0, acc01, false, false);
            acc01 = __builtin_amdgcn_wmma_f32_16x16x32_bf16(false, A0h, false, B1l, (short)0, acc01, false, false);
            acc01 = __builtin_amdgcn_wmma_f32_16x16x32_bf16(false, A0l, false, B1h, (short)0, acc01, false, false);

            acc10 = __builtin_amdgcn_wmma_f32_16x16x32_bf16(false, A1h, false, B0h, (short)0, acc10, false, false);
            acc10 = __builtin_amdgcn_wmma_f32_16x16x32_bf16(false, A1h, false, B0l, (short)0, acc10, false, false);
            acc10 = __builtin_amdgcn_wmma_f32_16x16x32_bf16(false, A1l, false, B0h, (short)0, acc10, false, false);

            acc11 = __builtin_amdgcn_wmma_f32_16x16x32_bf16(false, A1h, false, B1h, (short)0, acc11, false, false);
            acc11 = __builtin_amdgcn_wmma_f32_16x16x32_bf16(false, A1h, false, B1l, (short)0, acc11, false, false);
            acc11 = __builtin_amdgcn_wmma_f32_16x16x32_bf16(false, A1l, false, B1h, (short)0, acc11, false, false);
        }
        __syncthreads();   // protect bf16 tiles before next conversion
    }

    // ---- epilogue: D layout (ISA 7.12.2): lane<16 -> M=r, N=lane; lane>=16 -> M=r+8 ----
    const int mrow = (lane >> 4) << 3;
    const int pxb  = y0 * 64 + wp * 32 + n;    // pixels contiguous over the row pair
    const size_t base = ((size_t)b * CCH) << 12;
    #pragma unroll
    for (int r = 0; r < 8; ++r) {
        y[base + ((size_t)(ocb +  0 + r + mrow) << 12) + pxb     ] = acc00[r];
        y[base + ((size_t)(ocb +  0 + r + mrow) << 12) + pxb + 16] = acc01[r];
        y[base + ((size_t)(ocb + 16 + r + mrow) << 12) + pxb     ] = acc10[r];
        y[base + ((size_t)(ocb + 16 + r + mrow) << 12) + pxb + 16] = acc11[r];
    }
}

// ---------------------------------------------------------------------------
// Kernel 3: per-channel batch statistics (training-mode BN, population var).
// ---------------------------------------------------------------------------
__global__ void bn_stats(const float* __restrict__ y,
                         float* __restrict__ mean, float* __restrict__ rstd) {
    const int ch = blockIdx.x;
    const int tid = threadIdx.x;
    float s = 0.f, s2 = 0.f;
    for (int bimg = 0; bimg < BATCH; ++bimg) {
        const float4* p = (const float4*)(y + (((size_t)bimg * CCH + ch) << 12));
        for (int i = tid; i < HWSZ / 4; i += 256) {
            float4 v = p[i];
            s  += v.x + v.y + v.z + v.w;
            s2 += v.x * v.x + v.y * v.y + v.z * v.z + v.w * v.w;
        }
    }
    __shared__ float rs[256], rq[256];
    rs[tid] = s; rq[tid] = s2;
    __syncthreads();
    for (int d = 128; d > 0; d >>= 1) {
        if (tid < d) { rs[tid] += rs[tid + d]; rq[tid] += rq[tid + d]; }
        __syncthreads();
    }
    if (tid == 0) {
        const float inv_n = 1.f / (float)(BATCH * HWSZ);
        float m = rs[0] * inv_n;
        float var = rq[0] * inv_n - m * m;
        mean[ch] = m;
        rstd[ch] = rsqrtf(var + 1e-5f);
    }
}

// ---------------------------------------------------------------------------
// Kernel 4: normalize + affine + residual (in place on d_out), float4 wide.
// ---------------------------------------------------------------------------
__global__ void bn_apply(float4* __restrict__ y, const float4* __restrict__ x,
                         const float* __restrict__ mean, const float* __restrict__ rstd,
                         const float* __restrict__ g, const float* __restrict__ bb) {
    size_t i = (size_t)blockIdx.x * 256 + threadIdx.x;
    int ch = (int)((i >> 10) & 255);     // 1024 float4 per channel image
    float m  = mean[ch];
    float rg = rstd[ch] * g[ch];
    float be = bb[ch];
    float4 v = y[i];
    float4 o = x[i];
    float4 r;
    r.x = (v.x - m) * rg + be + o.x;
    r.y = (v.y - m) * rg + be + o.y;
    r.z = (v.z - m) * rg + be + o.z;
    r.w = (v.w - m) * rg + be + o.w;
    y[i] = r;
}

extern "C" void kernel_launch(void* const* d_in, const int* in_sizes, int n_in,
                              void* d_out, int out_size, void* d_ws, size_t ws_size,
                              hipStream_t stream) {
    const float* origin    = (const float*)d_in[0];
    // d_in[1..6]: attention branch — mathematically dead (softmax over a
    // size-1 axis yields all-ones), so weight = sum_e conv_para[e].
    const float* conv_para = (const float*)d_in[7];
    const float* norm_g    = (const float*)d_in[8];
    const float* norm_b    = (const float*)d_in[9];
    float* y = (float*)d_out;

    // workspace layout
    char* ws = (char*)d_ws;
    __bf16* awh  = (__bf16*)ws;                                  // 1,179,648 B
    __bf16* awl  = (__bf16*)(ws + (size_t)AFRAG_ELEMS * 2);      // 1,179,648 B
    float*  mean = (float*)(ws + (size_t)AFRAG_ELEMS * 4);       // 1 KB
    float*  rstd = mean + 256;

    prep_weight<<<AFRAG_ELEMS / 256, 256, 0, stream>>>(conv_para, awh, awl);
    conv_wmma<<<dim3(BATCH * 32, 4), 256, 0, stream>>>(origin, awh, awl, y);
    bn_stats<<<CCH, 256, 0, stream>>>(y, mean, rstd);
    bn_apply<<<(BATCH * CCH * HWSZ) / (4 * 256), 256, 0, stream>>>(
        (float4*)y, (const float4*)origin, mean, rstd, norm_g, norm_b);
}
